// HybridAttention_53068615909643
// MI455X (gfx1250) — matrix operand order repaired and checked by reading, not compiled, not run on hardware
//
#include <hip/hip_runtime.h>
#include <hip/hip_bf16.h>
#include <math.h>

// ---------------------------------------------------------------------------
// Types for CDNA5 WMMA (wave32)
// ---------------------------------------------------------------------------
typedef __bf16 bf16_t;
typedef __attribute__((ext_vector_type(16))) __bf16 v16bf;
typedef __attribute__((ext_vector_type(8)))  __bf16 v8bf;
typedef __attribute__((ext_vector_type(8)))  float  v8f;

// ---------------------------------------------------------------------------
// Problem constants (match reference)
// ---------------------------------------------------------------------------
constexpr int   B_    = 2;
constexpr int   L_    = 2048;
constexpr int   D_    = 1024;
constexpr int   NH    = 16;
constexpr int   KVH   = 4;
constexpr int   HD    = 64;
constexpr int   WIN   = 1024;
constexpr int   M_TOK = B_ * L_;          // 4096 tokens
constexpr float SCALE = 0.125f;           // 1/sqrt(64)
constexpr float NEGINF = -1e30f;

// ---------------------------------------------------------------------------
// WMMA helpers
// ---------------------------------------------------------------------------
__device__ __forceinline__ v8f wmma_bf16(v16bf a, v16bf b, v8f c) {
  // D = A(16x32) * B(32x16) + C, f32 accumulate
  return __builtin_amdgcn_wmma_f32_16x16x32_bf16(
      /*neg_a=*/false, a, /*neg_b=*/false, b,
      /*c_mod=*/(short)0, c, /*reuse_a=*/false, /*reuse_b=*/false);
}

// A-matrix lane layout (16x32 bf16): lane holds row (lane&15);
// element e -> K = 8*half + e  (e<8),  16 + 8*half + (e-8)  (e>=8)
__device__ __forceinline__ v16bf load_mat_a(const bf16_t* rowp, int half) {
  const int lo = half << 3;
  const v8bf x0 = *(const v8bf*)(rowp + lo);
  const v8bf x1 = *(const v8bf*)(rowp + 16 + lo);
  v16bf r;
#pragma unroll
  for (int e = 0; e < 8; ++e) { r[e] = x0[e]; r[e + 8] = x1[e]; }
  return r;
}

// B-matrix lane layout (32x16 bf16): lane holds column (lane&15);
// element e -> K = 8*half + e  (e<8),  16 + 8*half + (e-8)  (e>=8).  colp points at &Bcol[k0] (K-contiguous).
__device__ __forceinline__ v16bf load_mat_b(const bf16_t* colp, int half) {
  const int lo = half << 3;
  const v8bf x0 = *(const v8bf*)(colp + lo);
  const v8bf x1 = *(const v8bf*)(colp + 16 + lo);
  v16bf r;
#pragma unroll
  for (int e = 0; e < 8; ++e) { r[e] = x0[e]; r[e + 8] = x1[e]; }
  return r;
}

// Reductions across a 16-lane half of the wave (xor masks < 16 never cross
// the half boundary in wave32) — matches the C/D row layout.
__device__ __forceinline__ float red_max16(float v) {
#pragma unroll
  for (int m = 1; m < 16; m <<= 1) v = fmaxf(v, __shfl_xor(v, m, 32));
  return v;
}
__device__ __forceinline__ float red_add16(float v) {
#pragma unroll
  for (int m = 1; m < 16; m <<= 1) v += __shfl_xor(v, m, 32);
  return v;
}

// ---------------------------------------------------------------------------
// Elementwise prep kernels
// ---------------------------------------------------------------------------
__global__ __launch_bounds__(256) void cvt_f32_bf16(const float* __restrict__ in,
                                                    bf16_t* __restrict__ out, int n) {
  int i = blockIdx.x * blockDim.x + threadIdx.x;
  if (i < n) out[i] = (bf16_t)in[i];
}

// W[K][N] f32  ->  Wt[N][K] bf16  (so WMMA B columns are contiguous over K)
__global__ __launch_bounds__(256) void cvt_transpose_w(const float* __restrict__ in,
                                                       bf16_t* __restrict__ out,
                                                       int K, int N) {
  int i = blockIdx.x * blockDim.x + threadIdx.x;
  if (i >= K * N) return;
  int k = i / N, n = i % N;
  out[(size_t)n * K + k] = (bf16_t)in[i];
}

// RoPE + head split: in [B][L][nh*64] -> out [B][nh][L][64]
__global__ __launch_bounds__(256) void rope_kernel(const bf16_t* __restrict__ in,
                                                   bf16_t* __restrict__ out, int nh) {
  int i = blockIdx.x * blockDim.x + threadIdx.x;
  if (i >= B_ * L_ * nh * HD) return;
  const int d = i & 63;
  int t = i >> 6;
  const int h = t % nh;  t /= nh;
  const int l = t % L_;
  const int b = t / L_;

  const float x  = (float)in[i];
  const float xr = (d < 32) ? -(float)in[i + 32] : (float)in[i - 32];
  const int   fi = d & 31;
  // 1 / ROPE_BASE^(2*fi/64); ln(10000) = 9.210340371976184
  const float freq = __expf(-(2.0f * (float)fi / 64.0f) * 9.210340371976184f);
  const float ang  = (float)l * freq;
  float s, c;
  __sincosf(ang, &s, &c);
  out[(((size_t)b * nh + h) * L_ + l) * HD + d] = (bf16_t)(x * c + xr * s);
}

// V transpose: in [B][L][KVH*64] -> out [B][KVH][64][L]
__global__ __launch_bounds__(256) void vtrans_kernel(const bf16_t* __restrict__ in,
                                                     bf16_t* __restrict__ out) {
  int i = blockIdx.x * blockDim.x + threadIdx.x;
  if (i >= B_ * L_ * KVH * HD) return;
  const int d  = i & 63;
  const int kv = (i >> 6) & 3;
  int t = i >> 8;
  const int l = t % L_;
  const int b = t / L_;
  out[(((size_t)b * KVH + kv) * HD + d) * L_ + l] = in[i];
}

// ---------------------------------------------------------------------------
// WMMA GEMM:  Out[M][N] = A[M][K](bf16) * Bt[N][K](bf16)^T
// One wave -> one 32x64 macro-tile (2 M-tiles x 4 N-tiles, 8 accumulators):
// 8 WMMAs per 32-deep K-step off 12 b128 loads (1.5 loads/WMMA), and 8
// independent WMMA chains to cover XDL latency.  4 waves / block.
// Requires M % 32 == 0, N % 64 == 0, K % 32 == 0.
// ---------------------------------------------------------------------------
template <int OUTF32>
__global__ __launch_bounds__(128) void gemm_wmma(const bf16_t* __restrict__ A,
                                                 const bf16_t* __restrict__ Bt,
                                                 void* __restrict__ Out,
                                                 int M, int N, int K) {
  const int lane = threadIdx.x & 31;
  const int w    = threadIdx.x >> 5;
  const int ln   = lane & 15;
  const int half = lane >> 4;
  const int tile = blockIdx.x * 4 + w;
  const int ntn  = N >> 6;                 // 64-wide macro columns
  const int tn   = (tile % ntn) << 6;
  const int tm   = (tile / ntn) << 5;
  if (tm >= M) return;

  const bf16_t* ar0 = A  + (size_t)(tm +      ln) * K;
  const bf16_t* ar1 = A  + (size_t)(tm + 16 + ln) * K;
  const bf16_t* bc0 = Bt + (size_t)(tn +      ln) * K;
  const bf16_t* bc1 = Bt + (size_t)(tn + 16 + ln) * K;
  const bf16_t* bc2 = Bt + (size_t)(tn + 32 + ln) * K;
  const bf16_t* bc3 = Bt + (size_t)(tn + 48 + ln) * K;

  v8f acc[2][4] = {};
  for (int k0 = 0; k0 < K; k0 += 32) {
    __builtin_prefetch(ar0 + k0 + 256);    // global_prefetch_b8
    __builtin_prefetch(ar1 + k0 + 256);
    const v16bf a0 = load_mat_a(ar0 + k0, half);
    const v16bf a1 = load_mat_a(ar1 + k0, half);
    const v16bf b0 = load_mat_b(bc0 + k0, half);
    const v16bf b1 = load_mat_b(bc1 + k0, half);
    const v16bf b2 = load_mat_b(bc2 + k0, half);
    const v16bf b3 = load_mat_b(bc3 + k0, half);
    acc[0][0] = wmma_bf16(a0, b0, acc[0][0]);
    acc[0][1] = wmma_bf16(a0, b1, acc[0][1]);
    acc[0][2] = wmma_bf16(a0, b2, acc[0][2]);
    acc[0][3] = wmma_bf16(a0, b3, acc[0][3]);
    acc[1][0] = wmma_bf16(a1, b0, acc[1][0]);
    acc[1][1] = wmma_bf16(a1, b1, acc[1][1]);
    acc[1][2] = wmma_bf16(a1, b2, acc[1][2]);
    acc[1][3] = wmma_bf16(a1, b3, acc[1][3]);
  }
  if (OUTF32) {
    float* o = (float*)Out;
#pragma unroll
    for (int mi = 0; mi < 2; ++mi)
#pragma unroll
      for (int nj = 0; nj < 4; ++nj)
#pragma unroll
        for (int r = 0; r < 8; ++r)
          o[(size_t)(tm + mi * 16 + r + 8 * half) * N + tn + nj * 16 + ln] = acc[mi][nj][r];
  } else {
    bf16_t* o = (bf16_t*)Out;
#pragma unroll
    for (int mi = 0; mi < 2; ++mi)
#pragma unroll
      for (int nj = 0; nj < 4; ++nj)
#pragma unroll
        for (int r = 0; r < 8; ++r)
          o[(size_t)(tm + mi * 16 + r + 8 * half) * N + tn + nj * 16 + ln] = (bf16_t)acc[mi][nj][r];
  }
}

// ---------------------------------------------------------------------------
// Flash attention (sliding-window causal, GQA), one wave per 16-query tile.
//   qr [B][NH][L][64]   (RoPE'd, bf16)
//   kr [B][KVH][L][64]  (RoPE'd, bf16)
//   vt [B][KVH][64][L]  (bf16, key-contiguous)
//   ao [B][L][NH*64]    (bf16)
// ---------------------------------------------------------------------------
__global__ __launch_bounds__(128) void attn_kernel(const bf16_t* __restrict__ qr,
                                                   const bf16_t* __restrict__ kr,
                                                   const bf16_t* __restrict__ vt,
                                                   bf16_t* __restrict__ ao) {
  __shared__ __align__(16) bf16_t lds_p[4][16][40];  // per-wave P staging (padded)

  const int lane = threadIdx.x & 31;
  const int w    = threadIdx.x >> 5;
  const int ln   = lane & 15;
  const int half = lane >> 4;

  const int gw  = blockIdx.x * 4 + w;              // 0 .. B*NH*(L/16)-1
  const int mt  = gw % (L_ / 16);
  const int h   = (gw / (L_ / 16)) % NH;
  const int b   = gw / ((L_ / 16) * NH);
  const int q0  = mt * 16;
  const int kvh = h / (NH / KVH);

  // Q tile in WMMA-A layout (row = ln), two K-steps covering HD=64
  const bf16_t* qrow = qr + (((size_t)(b * NH + h)) * L_ + (q0 + ln)) * HD;
  const v16bf qa0 = load_mat_a(qrow,      half);
  const v16bf qa1 = load_mat_a(qrow + 32, half);

  const bf16_t* kbase = kr + ((size_t)(b * KVH + kvh)) * L_ * HD;
  const bf16_t* vbase = vt + ((size_t)(b * KVH + kvh)) * HD * (size_t)L_;

  float m_r[8], l_r[8];
  v8f o0 = {}, o1 = {}, o2 = {}, o3 = {};
#pragma unroll
  for (int r = 0; r < 8; ++r) { m_r[r] = NEGINF; l_r[r] = 0.0f; }

  int jlo = q0 - (WIN - 1);
  if (jlo < 0) jlo = 0;
  jlo &= ~31;                       // 32-key chunks; L % 32 == 0 -> no OOB
  const int jhi = q0 + 15;

  for (int j0 = jlo; j0 <= jhi; j0 += 32) {
    // ---- S = Q (16x64) * K^T (64x32): 4 WMMAs ----
    v8f s0 = {}, s1 = {};
    {
      const bf16_t* c0 = kbase + (size_t)(j0 + ln) * HD;        // key col nt=0
      s0 = wmma_bf16(qa0, load_mat_b(c0,      half), s0);
      s0 = wmma_bf16(qa1, load_mat_b(c0 + 32, half), s0);
      const bf16_t* c1 = kbase + (size_t)(j0 + 16 + ln) * HD;   // key col nt=1
      s1 = wmma_bf16(qa0, load_mat_b(c1,      half), s1);
      s1 = wmma_bf16(qa1, load_mat_b(c1 + 32, half), s1);
    }

    // ---- mask + online softmax (rows live in 16-lane halves) ----
#pragma unroll
    for (int r = 0; r < 8; ++r) {
      const int i  = q0 + r + 8 * half;
      const int jA = j0 + ln;
      const int jB = j0 + 16 + ln;
      float a = s0[r] * SCALE;
      float c = s1[r] * SCALE;
      if (!((jA <= i) && (i - jA < WIN))) a = NEGINF;
      if (!((jB <= i) && (i - jB < WIN))) c = NEGINF;
      const float mx    = red_max16(fmaxf(a, c));
      const float mnew  = fmaxf(m_r[r], mx);
      const float mb    = (mnew < -1e29f) ? 0.0f : mnew;  // NaN-safe bias
      const float alpha = __expf(m_r[r] - mb);
      const float p0    = __expf(a - mb);                 // masked -> exactly 0
      const float p1    = __expf(c - mb);
      const float rs    = red_add16(p0 + p1);
      l_r[r] = l_r[r] * alpha + rs;
      m_r[r] = mnew;
      o0[r] *= alpha; o1[r] *= alpha; o2[r] *= alpha; o3[r] *= alpha;
      lds_p[w][r + 8 * half][ln]      = (bf16_t)p0;       // C-layout -> LDS
      lds_p[w][r + 8 * half][ln + 16] = (bf16_t)p1;
    }
    // same-wave LDS RAW: DS ops are in-order per wave; wait + compiler fence
    asm volatile("s_wait_dscnt 0" ::: "memory");
    const v16bf pa = load_mat_a(&lds_p[w][ln][0], half);  // P in WMMA-A layout

    // ---- O += P (16x32) * V (32x64): 4 WMMAs ----
    o0 = wmma_bf16(pa, load_mat_b(vbase + (size_t)( 0 + ln) * L_ + j0, half), o0);
    o1 = wmma_bf16(pa, load_mat_b(vbase + (size_t)(16 + ln) * L_ + j0, half), o1);
    o2 = wmma_bf16(pa, load_mat_b(vbase + (size_t)(32 + ln) * L_ + j0, half), o2);
    o3 = wmma_bf16(pa, load_mat_b(vbase + (size_t)(48 + ln) * L_ + j0, half), o3);
    asm volatile("" ::: "memory");  // keep LDS loads ahead of next-iter stores
  }

  // ---- epilogue: normalize and scatter to [B][L][NH*HD] ----
#pragma unroll
  for (int r = 0; r < 8; ++r) {
    const float inv = 1.0f / l_r[r];
    const int   row = q0 + r + 8 * half;
    bf16_t* orow = ao + ((size_t)b * L_ + row) * (NH * HD) + h * HD;
    orow[ 0 + ln] = (bf16_t)(o0[r] * inv);
    orow[16 + ln] = (bf16_t)(o1[r] * inv);
    orow[32 + ln] = (bf16_t)(o2[r] * inv);
    orow[48 + ln] = (bf16_t)(o3[r] * inv);
  }
}

// ---------------------------------------------------------------------------
// Host launcher
// ---------------------------------------------------------------------------
extern "C" void kernel_launch(void* const* d_in, const int* in_sizes, int n_in,
                              void* d_out, int out_size, void* d_ws, size_t ws_size,
                              hipStream_t stream) {
  (void)in_sizes; (void)n_in; (void)out_size; (void)ws_size;
  const float* x  = (const float*)d_in[0];
  const float* Wq = (const float*)d_in[1];
  const float* Wk = (const float*)d_in[2];
  const float* Wv = (const float*)d_in[3];
  const float* Wo = (const float*)d_in[4];
  float* out = (float*)d_out;

  char*  ws  = (char*)d_ws;
  size_t off = 0;
  auto alloc = [&](size_t elems) -> bf16_t* {
    bf16_t* p = (bf16_t*)(ws + off);
    off += (elems * sizeof(bf16_t) + 255) & ~(size_t)255;
    return p;
  };
  bf16_t* xb    = alloc((size_t)M_TOK * D_);        // x in bf16
  bf16_t* wq_t  = alloc((size_t)1024 * D_);         // [N][K]
  bf16_t* wk_t  = alloc((size_t)256  * D_);
  bf16_t* wv_t  = alloc((size_t)256  * D_);
  bf16_t* wo_t  = alloc((size_t)1024 * 1024);
  bf16_t* q_raw = alloc((size_t)M_TOK * 1024);      // [B][L][NH*64]
  bf16_t* k_raw = alloc((size_t)M_TOK * 256);
  bf16_t* v_raw = alloc((size_t)M_TOK * 256);
  bf16_t* q_rp  = alloc((size_t)B_ * NH  * L_ * HD);   // [B][NH][L][64]
  bf16_t* k_rp  = alloc((size_t)B_ * KVH * L_ * HD);   // [B][KVH][L][64]
  bf16_t* v_t   = alloc((size_t)B_ * KVH * HD * L_);   // [B][KVH][64][L]
  bf16_t* a_out = alloc((size_t)M_TOK * 1024);         // [B][L][NH*64]

  // 1) convert inputs
  {
    int n = M_TOK * D_;
    cvt_f32_bf16<<<n / 256, 256, 0, stream>>>(x, xb, n);
    cvt_transpose_w<<<(D_ * 1024) / 256, 256, 0, stream>>>(Wq, wq_t, D_, 1024);
    cvt_transpose_w<<<(D_ * 256)  / 256, 256, 0, stream>>>(Wk, wk_t, D_, 256);
    cvt_transpose_w<<<(D_ * 256)  / 256, 256, 0, stream>>>(Wv, wv_t, D_, 256);
    cvt_transpose_w<<<(1024 * 1024) / 256, 256, 0, stream>>>(Wo, wo_t, 1024, 1024);
  }
  // 2) QKV projections (WMMA bf16, f32 accumulate); 32x64 macro-tiles, 4/block
  gemm_wmma<0><<<((M_TOK / 32) * (1024 / 64)) / 4, 128, 0, stream>>>(xb, wq_t, q_raw, M_TOK, 1024, D_);
  gemm_wmma<0><<<((M_TOK / 32) * (256  / 64)) / 4, 128, 0, stream>>>(xb, wk_t, k_raw, M_TOK, 256, D_);
  gemm_wmma<0><<<((M_TOK / 32) * (256  / 64)) / 4, 128, 0, stream>>>(xb, wv_t, v_raw, M_TOK, 256, D_);
  // 3) RoPE + layout, V transpose
  rope_kernel<<<(B_ * L_ * NH  * HD) / 256, 256, 0, stream>>>(q_raw, q_rp, NH);
  rope_kernel<<<(B_ * L_ * KVH * HD) / 256, 256, 0, stream>>>(k_raw, k_rp, KVH);
  vtrans_kernel<<<(B_ * L_ * KVH * HD) / 256, 256, 0, stream>>>(v_raw, v_t);
  // 4) sliding-window flash attention
  attn_kernel<<<(B_ * NH * (L_ / 16)) / 4, 128, 0, stream>>>(q_rp, k_rp, v_t, a_out);
  // 5) output projection -> f32 d_out
  gemm_wmma<1><<<((M_TOK / 32) * (1024 / 64)) / 4, 128, 0, stream>>>(a_out, wo_t, (void*)out, M_TOK, 1024, 1024);
}
